// SelfAttentionSN_45749991637367
// MI455X (gfx1250) — matrix operand order repaired and checked by reading, not compiled, not run on hardware
//
#include <hip/hip_runtime.h>

// ---------------------------------------------------------------------------
// Spectral-norm multi-head self-attention for MI455X (gfx1250, wave32).
// bf16 WMMA (v_wmma_f32_16x16x32_bf16) for all matmuls, fp32 accumulate,
// fp32 softmax held entirely in 320KB LDS so attn is written to HBM once.
// ---------------------------------------------------------------------------

typedef unsigned short u16;
typedef __attribute__((ext_vector_type(16))) __bf16 bf16x16;
typedef __attribute__((ext_vector_type(8)))  float  f32x8;

struct alignas(16) B16 { unsigned int x0, x1, x2, x3; };
union Frag { bf16x16 v; B16 h[2]; };

__device__ inline u16 f2bfu(float f) {          // round-to-nearest-even fp32->bf16
  unsigned u = __builtin_bit_cast(unsigned, f);
  u += 0x7FFFu + ((u >> 16) & 1u);
  return (u16)(u >> 16);
}
__device__ inline f32x8 zero8() {
  f32x8 z = {0.f,0.f,0.f,0.f,0.f,0.f,0.f,0.f};
  return z;
}
__device__ inline f32x8 wmma_bf16(const Frag& a, const Frag& b, f32x8 c) {
  return __builtin_amdgcn_wmma_f32_16x16x32_bf16(false, a.v, false, b.v,
                                                 (short)0, c, false, false);
}

// ---------------------------------------------------------------------------
// Kernel 1: inverse spectral norm. sigma = ||v @ W|| / ||v||, v = u @ W^T.
// One block per weight matrix (4 blocks x 256 threads). W is [1024 x 1024].
// ---------------------------------------------------------------------------
__global__ __launch_bounds__(256) void sigma_kernel(
    const float* __restrict__ w0, const float* __restrict__ u0,
    const float* __restrict__ w1, const float* __restrict__ u1,
    const float* __restrict__ w2, const float* __restrict__ u2,
    const float* __restrict__ w3, const float* __restrict__ u3,
    float* __restrict__ inv_sigma)
{
  const int F = 1024;
  int m = blockIdx.x;
  const float* W = (m == 0) ? w0 : (m == 1) ? w1 : (m == 2) ? w2 : w3;
  const float* U = (m == 0) ? u0 : (m == 1) ? u1 : (m == 2) ? u2 : u3;
  __shared__ float us[1024];
  __shared__ float vs[1024];
  __shared__ float red[256];
  int tid = threadIdx.x;
  for (int i = tid; i < F; i += 256) us[i] = U[i];
  __syncthreads();
  // v[i] = sum_o u[o] * W[i, o]
  float v2 = 0.f;
  for (int i = tid; i < F; i += 256) {
    const float* row = W + (size_t)i * F;
    float acc = 0.f;
    for (int o = 0; o < F; ++o) acc += us[o] * row[o];
    vs[i] = acc;
    v2 += acc * acc;
  }
  red[tid] = v2;
  __syncthreads();
  for (int s = 128; s > 0; s >>= 1) {
    if (tid < s) red[tid] += red[tid + s];
    __syncthreads();
  }
  float vnorm2 = red[0];
  __syncthreads();
  // un[o] = sum_i v[i] * W[i, o];  sigma = ||un|| / ||v||
  float sq = 0.f;
  for (int o = tid; o < F; o += 256) {
    float acc = 0.f;
    for (int i = 0; i < F; ++i) acc += vs[i] * W[(size_t)i * F + o];
    sq += acc * acc;
  }
  red[tid] = sq;
  __syncthreads();
  for (int s = 128; s > 0; s >>= 1) {
    if (tid < s) red[tid] += red[tid + s];
    __syncthreads();
  }
  if (tid == 0) inv_sigma[m] = sqrtf(vnorm2) / sqrtf(red[0]);
}

// ---------------------------------------------------------------------------
// Kernel 2: tiled WMMA GEMM  C[8192,1024] = A[8192,1024] @ (W * inv_sigma) + b
// Block tile 128x128, BK=32, 8 waves in a 2x4 grid, 64x32 per wave (8 WMMAs).
// MODE 0: A=fp32 x, out bf16 [B,H,S,D]   (Q / K projections)
// MODE 2: A=fp32 x, out bf16 [B,H,D,S]   (V projection, pre-transposed)
// MODE 3: A=bf16 ctx, out fp32 = x + (C + b) * gamma   (output projection)
// ---------------------------------------------------------------------------
template <int MODE>
__global__ __launch_bounds__(256) void gemm_kernel(
    const float* __restrict__ Xf, const u16* __restrict__ Xb,
    const float* __restrict__ W,  const float* __restrict__ Bias,
    const float* __restrict__ InvSig, int sidx,
    const float* __restrict__ Xres, const float* __restrict__ Gamma,
    u16* __restrict__ OutB, float* __restrict__ OutF)
{
  __shared__ u16 As[128 * 40];   // [row][k] padded to 40 (80B rows, 16B aligned)
  __shared__ u16 Bs[128 * 40];   // [n][k] transposed, padded to 40
  int tid = threadIdx.x;
  int m0 = blockIdx.x * 128, n0 = blockIdx.y * 128;
  int lane = tid & 31, wid = tid >> 5, lh = lane & 15, hi = lane >> 4;
  int wm = wid >> 2, wn = wid & 3;
  float is = InvSig[sidx];
  f32x8 acc[4][2];
  #pragma unroll
  for (int i = 0; i < 4; ++i) { acc[i][0] = zero8(); acc[i][1] = zero8(); }

  int tr = tid >> 3;            // 0..31 (row group)
  int tc = (tid & 7) * 4;       // A: 4 cols each
  int bn0 = (tid & 7) * 16;     // B: 16 n's each

  for (int kt = 0; kt < 32; ++kt) {
    int k0 = kt * 32;
    // ---- stage A tile (128x32) ----
    #pragma unroll
    for (int p = 0; p < 4; ++p) {
      int row = tr + p * 32;
      u16* dp = As + row * 40 + tc;
      if constexpr (MODE == 3) {
        const u16* gp = Xb + (size_t)(m0 + row) * 1024 + k0 + tc;
        dp[0] = gp[0]; dp[1] = gp[1]; dp[2] = gp[2]; dp[3] = gp[3];
      } else {
        const float4 f = *(const float4*)(Xf + (size_t)(m0 + row) * 1024 + k0 + tc);
        dp[0] = f2bfu(f.x); dp[1] = f2bfu(f.y); dp[2] = f2bfu(f.z); dp[3] = f2bfu(f.w);
      }
    }
    // ---- stage B tile transposed (Bs[n][k]), fused * inv_sigma ----
    #pragma unroll
    for (int p = 0; p < 4; ++p) {
      int n = bn0 + p * 4;
      const float4 f = *(const float4*)(W + (size_t)(k0 + tr) * 1024 + n0 + n);
      Bs[(n + 0) * 40 + tr] = f2bfu(f.x * is);
      Bs[(n + 1) * 40 + tr] = f2bfu(f.y * is);
      Bs[(n + 2) * 40 + tr] = f2bfu(f.z * is);
      Bs[(n + 3) * 40 + tr] = f2bfu(f.w * is);
    }
    __syncthreads();
    // ---- fragments + 8 WMMAs ----
    Frag fb[2];
    #pragma unroll
    for (int nt = 0; nt < 2; ++nt) {
      const u16* bp = Bs + (wn * 32 + nt * 16 + lh) * 40 + hi * 8;
      fb[nt].h[0] = *(const B16*)bp;
      fb[nt].h[1] = *(const B16*)(bp + 16);
    }
    #pragma unroll
    for (int mt = 0; mt < 4; ++mt) {
      Frag fa;
      const u16* ap = As + (wm * 64 + mt * 16 + lh) * 40;
      fa.h[0] = *(const B16*)(ap + hi * 8);
      fa.h[1] = *(const B16*)(ap + 16 + hi * 8);
      acc[mt][0] = wmma_bf16(fa, fb[0], acc[mt][0]);
      acc[mt][1] = wmma_bf16(fa, fb[1], acc[mt][1]);
    }
    __syncthreads();
  }
  // ---- epilogue ----
  float g = (MODE == 3) ? Gamma[0] : 0.f;
  #pragma unroll
  for (int nt = 0; nt < 2; ++nt) {
    int n = n0 + wn * 32 + nt * 16 + lh;
    float bv = Bias[n];
    #pragma unroll
    for (int mt = 0; mt < 4; ++mt) {
      #pragma unroll
      for (int r = 0; r < 8; ++r) {
        int m = m0 + wm * 64 + mt * 16 + hi * 8 + r;
        float val = acc[mt][nt][r] + bv;
        if constexpr (MODE == 3) {
          OutF[(size_t)m * 1024 + n] = Xres[(size_t)m * 1024 + n] + val * g;
        } else {
          int b = m >> 11, s = m & 2047, h = n >> 7, d = n & 127;
          size_t a;
          if constexpr (MODE == 2)
            a = ((size_t)((b * 8 + h) * 128 + d)) * 2048 + s;   // [B,H,D,S]
          else
            a = ((size_t)((b * 8 + h) * 2048 + s)) * 128 + d;   // [B,H,S,D]
          OutB[a] = f2bfu(val);
        }
      }
    }
  }
}

// ---------------------------------------------------------------------------
// Kernel 3: fused attention. One block = one (b,h) and 32 query rows.
// Phase 1: logits = Q@K^T (WMMA) into 32x2049 fp32 LDS slab (~256KB).
// Phase 2: exact fp32 softmax stats in LDS.
// Phase 3: single normalized fp32 attn write to HBM + bf16 P staging +
//          ctx = P@V (WMMA, V pre-transposed [B,H,D,S]).
// ---------------------------------------------------------------------------
__global__ __launch_bounds__(256) void attn_kernel(
    const u16* __restrict__ Qh, const u16* __restrict__ Kh,
    const u16* __restrict__ Vt, float* __restrict__ Attn,
    u16* __restrict__ Ctx)
{
  extern __shared__ char smem[];
  const int LS = 2049;                    // padded logits row stride (floats)
  const int PW = 136;                     // padded P row stride (bf16)
  float* Ld    = (float*)smem;                                  // 32*2049 f32
  u16*   Pst   = (u16*)(smem + 32 * LS * 4);                    // 32*136 bf16
  float* rowinv = (float*)(smem + 32 * LS * 4 + 32 * PW * 2);   // 32
  float* rowmax = rowinv + 32;                                  // 32
  float* red    = rowmax + 32;                                  // 256

  int tid = threadIdx.x, lane = tid & 31, wid = tid >> 5;
  int lh = lane & 15, hi = lane >> 4;
  int blk = blockIdx.x;
  int bh = blk >> 6;                      // (b*8+h)
  int m0 = (blk & 63) * 32;
  const u16* Qb = Qh + (size_t)bh * 2048 * 128;
  const u16* Kb = Kh + (size_t)bh * 2048 * 128;
  const u16* Vb = Vt + (size_t)bh * 128 * 2048;

  int wm = wid >> 2;                      // 0..1 : 16-row tile
  int wn = wid & 3;                       // 0..3 : 32-col group

  // ---- Phase 1: Q fragments live in registers for the whole sweep ----
  Frag fq[4];
  {
    const u16* qrow = Qb + (size_t)(m0 + wm * 16 + lh) * 128;
    #pragma unroll
    for (int kk = 0; kk < 4; ++kk) {
      int d0 = kk * 32;
      fq[kk].h[0] = *(const B16*)(qrow + d0 + hi * 8);
      fq[kk].h[1] = *(const B16*)(qrow + d0 + 16 + hi * 8);
    }
  }
  for (int c = 0; c < 16; ++c) {
    int n0c = c * 128;
    f32x8 s0 = zero8(), s1 = zero8();
    #pragma unroll
    for (int kk = 0; kk < 4; ++kk) {
      int d0 = kk * 32;
      Frag fk0, fk1;
      const u16* kr0 = Kb + (size_t)(n0c + wn * 32 + lh) * 128 + d0 + hi * 8;
      fk0.h[0] = *(const B16*)kr0;
      fk0.h[1] = *(const B16*)(kr0 + 16);
      const u16* kr1 = kr0 + 16 * 128;
      fk1.h[0] = *(const B16*)kr1;
      fk1.h[1] = *(const B16*)(kr1 + 16);
      s0 = wmma_bf16(fq[kk], fk0, s0);
      s1 = wmma_bf16(fq[kk], fk1, s1);
    }
    int colb = n0c + wn * 32;
    #pragma unroll
    for (int r = 0; r < 8; ++r) {
      int ml = wm * 16 + hi * 8 + r;
      Ld[ml * LS + colb + lh]      = s0[r];
      Ld[ml * LS + colb + 16 + lh] = s1[r];
    }
  }
  __syncthreads();

  // ---- Phase 2: rowmax, exp, rowsum (each lane owns a distinct row) ----
  int rr = tid & 31, part = tid >> 5;     // 8 segments of 256 cols per row
  float* lrow = Ld + rr * LS + part * 256;
  float mx = -3.402823e38f;
  for (int i = 0; i < 256; ++i) mx = fmaxf(mx, lrow[i]);
  red[part * 32 + rr] = mx;
  __syncthreads();
  if (tid < 32) {
    float v = red[tid];
    #pragma unroll
    for (int p = 1; p < 8; ++p) v = fmaxf(v, red[p * 32 + tid]);
    rowmax[tid] = v;
  }
  __syncthreads();
  float rm = rowmax[rr];
  float sum = 0.f;
  for (int i = 0; i < 256; ++i) {
    float e = __expf(lrow[i] - rm);
    lrow[i] = e;
    sum += e;
  }
  red[part * 32 + rr] = sum;
  __syncthreads();
  if (tid < 32) {
    float v = red[tid];
    #pragma unroll
    for (int p = 1; p < 8; ++p) v += red[p * 32 + tid];
    rowinv[tid] = 1.0f / v;
  }
  __syncthreads();

  // ---- Phase 3: write attn once + ctx = P @ V ----
  f32x8 c0 = zero8(), c1 = zero8();
  int wd = wid & 3;                       // 0..3 : 32-wide d group
  int tr3 = tid >> 3;                     // row 0..31
  int tc3 = (tid & 7) * 16;               // 16 cols each
  float* attn_row = Attn + ((size_t)bh * 2048 + m0 + tr3) * 2048;
  for (int c = 0; c < 16; ++c) {
    int n0c = c * 128;
    float inv = rowinv[tr3];
    const float* src = Ld + tr3 * LS + n0c + tc3;
    u16* pdst = Pst + tr3 * PW + tc3;
    #pragma unroll
    for (int j = 0; j < 16; j += 4) {
      float4 p;
      p.x = src[j] * inv; p.y = src[j + 1] * inv;
      p.z = src[j + 2] * inv; p.w = src[j + 3] * inv;
      *(float4*)(attn_row + n0c + tc3 + j) = p;
      pdst[j]     = f2bfu(p.x); pdst[j + 1] = f2bfu(p.y);
      pdst[j + 2] = f2bfu(p.z); pdst[j + 3] = f2bfu(p.w);
    }
    __syncthreads();
    #pragma unroll
    for (int kk = 0; kk < 4; ++kk) {
      Frag fp;
      const u16* prow = Pst + (wm * 16 + lh) * PW + kk * 32;
      fp.h[0] = *(const B16*)(prow + hi * 8);
      fp.h[1] = *(const B16*)(prow + 16 + hi * 8);
      Frag fv0, fv1;
      const u16* vr0 = Vb + (size_t)(wd * 32 + lh) * 2048 + n0c + kk * 32 + hi * 8;
      fv0.h[0] = *(const B16*)vr0;
      fv0.h[1] = *(const B16*)(vr0 + 16);
      const u16* vr1 = vr0 + 16 * 2048;
      fv1.h[0] = *(const B16*)vr1;
      fv1.h[1] = *(const B16*)(vr1 + 16);
      c0 = wmma_bf16(fp, fv0, c0);
      c1 = wmma_bf16(fp, fv1, c1);
    }
    __syncthreads();
  }
  // ctx epilogue -> [B,S,F] bf16 (concat-heads layout for the output GEMM)
  int b = bh >> 3, h = bh & 7;
  size_t cb = ((size_t)b * 2048) * 1024 + h * 128;
  #pragma unroll
  for (int r = 0; r < 8; ++r) {
    int s = m0 + wm * 16 + hi * 8 + r;
    int d0t = wd * 32;
    Ctx[cb + (size_t)s * 1024 + d0t + lh]      = f2bfu(c0[r]);
    Ctx[cb + (size_t)s * 1024 + d0t + 16 + lh] = f2bfu(c1[r]);
  }
}

// ---------------------------------------------------------------------------
// Host launcher
// ---------------------------------------------------------------------------
extern "C" void kernel_launch(void* const* d_in, const int* in_sizes, int n_in,
                              void* d_out, int out_size, void* d_ws, size_t ws_size,
                              hipStream_t stream) {
  (void)in_sizes; (void)n_in; (void)out_size; (void)ws_size;
  const float* x  = (const float*)d_in[0];
  const float* wq = (const float*)d_in[1];
  const float* bq = (const float*)d_in[2];
  const float* uq = (const float*)d_in[3];
  const float* wk = (const float*)d_in[4];
  const float* bk = (const float*)d_in[5];
  const float* uk = (const float*)d_in[6];
  const float* wv = (const float*)d_in[7];
  const float* bv = (const float*)d_in[8];
  const float* uv = (const float*)d_in[9];
  const float* wo = (const float*)d_in[10];
  const float* bo = (const float*)d_in[11];
  const float* uo = (const float*)d_in[12];
  const float* gamma = (const float*)d_in[13];

  float* outF  = (float*)d_out;                       // [4,2048,1024]
  float* attnF = outF + (size_t)8388608;              // [4,8,2048,2048]

  char* ws = (char*)d_ws;
  float* invsig = (float*)ws;                         // 4 floats
  u16* Qh  = (u16*)(ws + 256);                        // [B,H,S,D] bf16, 16MB
  u16* Kh  = Qh + (size_t)8388608;                    // [B,H,S,D] bf16
  u16* Vt  = Kh + (size_t)8388608;                    // [B,H,D,S] bf16
  u16* Ctx = Vt + (size_t)8388608;                    // [B,S,F]   bf16

  sigma_kernel<<<4, 256, 0, stream>>>(wq, uq, wk, uk, wv, uv, wo, uo, invsig);

  dim3 pg(64, 8);
  gemm_kernel<0><<<pg, 256, 0, stream>>>(x, nullptr, wq, bq, invsig, 0,
                                         nullptr, nullptr, Qh, nullptr);
  gemm_kernel<0><<<pg, 256, 0, stream>>>(x, nullptr, wk, bk, invsig, 1,
                                         nullptr, nullptr, Kh, nullptr);
  gemm_kernel<2><<<pg, 256, 0, stream>>>(x, nullptr, wv, bv, invsig, 2,
                                         nullptr, nullptr, Vt, nullptr);

  size_t smem = (size_t)32 * 2049 * 4 + (size_t)32 * 136 * 2 + 64 * 4 + 256 * 4;
  attn_kernel<<<2048, 256, smem, stream>>>(Qh, Kh, Vt, attnF, Ctx);

  gemm_kernel<3><<<pg, 256, 0, stream>>>(nullptr, Ctx, wo, bo, invsig, 3,
                                         x, gamma, nullptr, outF);
}
